// SimpleRNN_46084999086228
// MI455X (gfx1250) — matrix operand order repaired and checked by reading, not compiled, not run on hardware
//
#include <hip/hip_runtime.h>
#include <math.h>

// ---------------------------------------------------------------------------
// SimpleRNN on MI455X (gfx1250): f16 WMMA GEMMs, f32 accumulation, LDS-staged
// tiles via gfx1250 async global->LDS copies (double buffered).
//   xp = x @ W_ih^T + b_ih + b_hh          (once, fused h1 = tanh(xp))
//   h  = tanh(xp + h @ W_hh^T)             (31 chained step kernels)
//   out = h @ W_head^T + b_head
// ---------------------------------------------------------------------------

typedef __attribute__((ext_vector_type(16))) _Float16 v16h;
typedef __attribute__((ext_vector_type(8)))  _Float16 v8h;
typedef __attribute__((ext_vector_type(4)))  _Float16 v4h;
typedef __attribute__((ext_vector_type(8)))  float    v8f;
typedef __attribute__((vector_size(16)))     int      vs4i;   // matches builtin proto

#define B_DIM   8192
#define IN_DIM  512
#define H_DIM   1024
#define OUT_DIM 512

// Tile geometry: 256 threads = 8 waves (2 M x 4 N), wave tile 64x64 (4x4 WMMA)
#define BM 128
#define BN 256
#define BK 32
#define LDS_STRIDE 40   // 32 halves + 8 pad -> 80B row stride, conflict-free b128

// ---------------------------------------------------------------------------
// Cheap tanh: prefer gfx1250 v_tanh_f32; else exp2+rcp (5 VALU ops).
// ---------------------------------------------------------------------------
__device__ __forceinline__ float fast_tanh(float x) {
#if __has_builtin(__builtin_amdgcn_tanhf)
    return __builtin_amdgcn_tanhf(x);
#else
    x = fminf(fmaxf(x, -20.0f), 20.0f);
    float t = __builtin_amdgcn_exp2f(2.885390081777927f * x);  // e^(2x)
    return __builtin_fmaf(-2.0f, __builtin_amdgcn_rcpf(t + 1.0f), 1.0f);
#endif
}

// ---------------------------------------------------------------------------
// Async global -> LDS 16B copy (gfx1250 GLOBAL_LOAD_ASYNC_TO_LDS_B128),
// with synchronous load+ds_store fallback if the builtin is absent.
// Builtin proto (from hipcc diagnostic): arg0 = vs4i* (global src),
// arg1 = vs4i* (LDS dst), then imm offset, imm cpol.
// ---------------------------------------------------------------------------
#if __has_builtin(__builtin_amdgcn_global_load_async_to_lds_b128)
#define ASYNC_STAGE 1
__device__ __forceinline__ void cp16_g2l(_Float16* lds, const _Float16* g) {
    __builtin_amdgcn_global_load_async_to_lds_b128(
        (__attribute__((address_space(1))) vs4i*)g,
        (__attribute__((address_space(3))) vs4i*)lds,
        0, 0);
}
#else
#define ASYNC_STAGE 0
__device__ __forceinline__ void cp16_g2l(_Float16* lds, const _Float16* g) {
    *(uint4*)lds = *(const uint4*)g;   // global_load_b128 + ds_store_b128
}
#endif

template <int N>
__device__ __forceinline__ void wait_async() {
#if ASYNC_STAGE
#if __has_builtin(__builtin_amdgcn_s_wait_asynccnt)
    __builtin_amdgcn_s_wait_asynccnt(N);
#else
    asm volatile("s_wait_asynccnt %0" :: "i"(N) : "memory");
#endif
#endif
}

// ---------------------------------------------------------------------------
// Stage one K-slab: A tile BMx32 and B tile BNx32 into (padded) LDS.
// 256 threads: 2 chunks/thread for A, 4 chunks/thread for B (6 asyncs/thread).
// ---------------------------------------------------------------------------
__device__ __forceinline__ void stage_tiles(
    _Float16* __restrict__ sAbuf, _Float16* __restrict__ sBbuf,
    const _Float16* __restrict__ A, const _Float16* __restrict__ Bt,
    int K, int m0blk, int n0blk, int k0)
{
    const int t = threadIdx.x;                 // 0..255
    {   // A: 128 rows x 32 halves; thread -> row t>>1, half-offset (t&1)*16
        const int row = t >> 1;
        const int c   = (t & 1) * 16;
        const _Float16* g = A + (size_t)(m0blk + row) * K + k0 + c;
        _Float16* l = sAbuf + row * LDS_STRIDE + c;
        cp16_g2l(l,     g);
        cp16_g2l(l + 8, g + 8);
    }
    {   // B: 256 rows x 32 halves; thread -> row t (64B)
        const _Float16* g = Bt + (size_t)(n0blk + t) * K + k0;
        _Float16* l = sBbuf + t * LDS_STRIDE;
        cp16_g2l(l,      g);
        cp16_g2l(l + 8,  g + 8);
        cp16_g2l(l + 16, g + 16);
        cp16_g2l(l + 24, g + 24);
    }
}

// ---------------------------------------------------------------------------
// Per-wave 4x4 WMMA tiles from staged LDS slabs.
// Fragment layouts per CDNA5 wave32 spec (CDNA5 ISA §7.12.2):
//   A 16x32 f16: lane m -> row m, K hi*8..+7 and hi*8+16..+23
//   B 32x16 f16: lane n -> col n, K hi*8..+7 and hi*8+16..+23
// ---------------------------------------------------------------------------
__device__ __forceinline__ void wmma_tiles_from_lds(
    const _Float16* __restrict__ sAbuf, const _Float16* __restrict__ sBbuf,
    v8f (&acc)[4][4])
{
    const int lane = threadIdx.x & 31;
    const int wave = threadIdx.x >> 5;
    const int wm = wave & 1, wn = wave >> 1;
    const int lr = lane & 15, hi = lane >> 4;

    v16h a[4], b[4];
#pragma unroll
    for (int i = 0; i < 4; ++i) {
        const _Float16* pa = sAbuf + (wm * 64 + i * 16 + lr) * LDS_STRIDE + hi * 8;
        union { v16h v; v8h h[2]; } u;
        u.h[0] = *(const v8h*)(pa);
        u.h[1] = *(const v8h*)(pa + 16);
        a[i] = u.v;
        const _Float16* pb = sBbuf + (wn * 64 + i * 16 + lr) * LDS_STRIDE + hi * 8;
        union { v16h v; v8h h[2]; } w;
        w.h[0] = *(const v8h*)(pb);
        w.h[1] = *(const v8h*)(pb + 16);
        b[i] = w.v;
    }
#pragma unroll
    for (int i = 0; i < 4; ++i)
#pragma unroll
        for (int j = 0; j < 4; ++j)
            acc[i][j] = __builtin_amdgcn_wmma_f32_16x16x32_f16(
                false, a[i], false, b[j], (short)0, acc[i][j], false, false);
}

// ---------------------------------------------------------------------------
// Full 128x256 block GEMM: C = A[M,K] * Bt[N,K]^T with double-buffered LDS.
// ---------------------------------------------------------------------------
__device__ __forceinline__ void gemm_128x256(
    const _Float16* __restrict__ A, const _Float16* __restrict__ Bt,
    int K, v8f (&acc)[4][4])
{
    __shared__ __align__(16) _Float16 sA[2][BM * LDS_STRIDE];
    __shared__ __align__(16) _Float16 sB[2][BN * LDS_STRIDE];

    const int m0blk = blockIdx.y * BM;
    const int n0blk = blockIdx.x * BN;

    const v8f vzero = {};
#pragma unroll
    for (int i = 0; i < 4; ++i)
#pragma unroll
        for (int j = 0; j < 4; ++j)
            acc[i][j] = vzero;

    const int nk = K >> 5;
    stage_tiles(sA[0], sB[0], A, Bt, K, m0blk, n0blk, 0);
    for (int k = 0; k < nk; ++k) {
        if (k + 1 < nk) {
            stage_tiles(sA[(k + 1) & 1], sB[(k + 1) & 1], A, Bt, K,
                        m0blk, n0blk, (k + 1) * BK);
            wait_async<6>();   // in-order: previous slab's 6 copies done
        } else {
            wait_async<0>();
        }
        __syncthreads();
        wmma_tiles_from_lds(sA[k & 1], sB[k & 1], acc);
        __syncthreads();       // protect buffer about to be refilled
    }
}

__device__ __forceinline__ void tile_coords(int& m0, int& n0) {
    const int wave = threadIdx.x >> 5;
    m0 = blockIdx.y * BM + (wave & 1) * 64;
    n0 = blockIdx.x * BN + (wave >> 1) * 64;
}

// C/D f32 16x16 layout: VGPR r, lanes 0-15 -> M=r, N=lane; lanes 16-31 -> M=8+r.

// ---------------------------------------------------------------------------
// f32 -> f16 conversion (vectorized x4)
// ---------------------------------------------------------------------------
__global__ void cvt_f32_to_f16(const float* __restrict__ src,
                               _Float16* __restrict__ dst, int n4) {
    int i = blockIdx.x * blockDim.x + threadIdx.x;
    if (i < n4) {
        float4 v = ((const float4*)src)[i];
        v4h o = { (_Float16)v.x, (_Float16)v.y, (_Float16)v.z, (_Float16)v.w };
        ((v4h*)dst)[i] = o;
    }
}

// ---------------------------------------------------------------------------
// xp kernel: xp = x16 @ Wih16^T + (b_ih + b_hh);  h1 = tanh(xp)
// ---------------------------------------------------------------------------
__global__ __launch_bounds__(256)
void rnn_xp_kernel(const _Float16* __restrict__ x16,
                   const _Float16* __restrict__ Wih16,
                   const float* __restrict__ b_ih,
                   const float* __restrict__ b_hh,
                   float* __restrict__ xp,
                   _Float16* __restrict__ h1)
{
    v8f acc[4][4];
    gemm_128x256(x16, Wih16, IN_DIM, acc);

    int m0, n0; tile_coords(m0, n0);
    const int lane = threadIdx.x & 31;
    const int lr = lane & 15, hi = lane >> 4;
#pragma unroll
    for (int i = 0; i < 4; ++i) {
#pragma unroll
        for (int j = 0; j < 4; ++j) {
            const int row = m0 + i * 16 + hi * 8;
            const int col = n0 + j * 16 + lr;
            const float bias = b_ih[col] + b_hh[col];
#pragma unroll
            for (int r = 0; r < 8; ++r) {
                float v = acc[i][j][r] + bias;
                xp[(size_t)(row + r) * H_DIM + col] = v;
                h1[(size_t)(row + r) * H_DIM + col] = (_Float16)fast_tanh(v);
            }
        }
    }
}

// ---------------------------------------------------------------------------
// Step kernel: h_out = tanh(xp + h_in @ Whh16^T)
// ---------------------------------------------------------------------------
__global__ __launch_bounds__(256)
void rnn_step_kernel(const _Float16* __restrict__ h_in,
                     const _Float16* __restrict__ Whh16,
                     const float* __restrict__ xp,
                     _Float16* __restrict__ h_out)
{
    v8f acc[4][4];
    gemm_128x256(h_in, Whh16, H_DIM, acc);

    int m0, n0; tile_coords(m0, n0);
    const int lane = threadIdx.x & 31;
    const int lr = lane & 15, hi = lane >> 4;
#pragma unroll
    for (int i = 0; i < 4; ++i) {
#pragma unroll
        for (int j = 0; j < 4; ++j) {
            const int row = m0 + i * 16 + hi * 8;
            const int col = n0 + j * 16 + lr;
#pragma unroll
            for (int r = 0; r < 8; ++r) {
                float v = fast_tanh(xp[(size_t)(row + r) * H_DIM + col] + acc[i][j][r]);
                h_out[(size_t)(row + r) * H_DIM + col] = (_Float16)v;
            }
        }
    }
}

// ---------------------------------------------------------------------------
// Head kernel: out = h @ Whead16^T + b_head   (out f32 [B, OUT_DIM])
// ---------------------------------------------------------------------------
__global__ __launch_bounds__(256)
void rnn_head_kernel(const _Float16* __restrict__ h16,
                     const _Float16* __restrict__ Whead16,
                     const float* __restrict__ b_head,
                     float* __restrict__ out)
{
    v8f acc[4][4];
    gemm_128x256(h16, Whead16, H_DIM, acc);

    int m0, n0; tile_coords(m0, n0);
    const int lane = threadIdx.x & 31;
    const int lr = lane & 15, hi = lane >> 4;
#pragma unroll
    for (int i = 0; i < 4; ++i) {
#pragma unroll
        for (int j = 0; j < 4; ++j) {
            const int row = m0 + i * 16 + hi * 8;
            const int col = n0 + j * 16 + lr;
            const float bias = b_head[col];
#pragma unroll
            for (int r = 0; r < 8; ++r)
                out[(size_t)(row + r) * OUT_DIM + col] = acc[i][j][r] + bias;
        }
    }
}

// ---------------------------------------------------------------------------
// Launch sequence (graph-capture safe: only kernel launches on `stream`)
// ---------------------------------------------------------------------------
extern "C" void kernel_launch(void* const* d_in, const int* in_sizes, int n_in,
                              void* d_out, int out_size, void* d_ws, size_t ws_size,
                              hipStream_t stream) {
    (void)in_sizes; (void)n_in; (void)out_size; (void)ws_size;

    const float* x      = (const float*)d_in[0];
    const float* W_ih   = (const float*)d_in[1];
    const float* b_ih   = (const float*)d_in[2];
    const float* W_hh   = (const float*)d_in[3];
    const float* b_hh   = (const float*)d_in[4];
    const float* W_head = (const float*)d_in[5];
    const float* b_head = (const float*)d_in[6];
    const int steps = 32;  // reference STEPS (d_in[7] is device-resident scalar)

    // Workspace layout (~76 MB)
    char* ws = (char*)d_ws;
    _Float16* x16     = (_Float16*)ws; ws += (size_t)B_DIM * IN_DIM * 2;
    _Float16* Wih16   = (_Float16*)ws; ws += (size_t)H_DIM * IN_DIM * 2;
    _Float16* Whh16   = (_Float16*)ws; ws += (size_t)H_DIM * H_DIM * 2;
    _Float16* Whead16 = (_Float16*)ws; ws += (size_t)OUT_DIM * H_DIM * 2;
    float*    xp      = (float*)ws;    ws += (size_t)B_DIM * H_DIM * 4;
    _Float16* hA      = (_Float16*)ws; ws += (size_t)B_DIM * H_DIM * 2;
    _Float16* hB      = (_Float16*)ws;

    auto cvt = [&](const float* s, _Float16* d, size_t n) {
        int n4 = (int)(n / 4);
        cvt_f32_to_f16<<<dim3((n4 + 255) / 256), dim3(256), 0, stream>>>(s, d, n4);
    };
    cvt(x,      x16,     (size_t)B_DIM * IN_DIM);
    cvt(W_ih,   Wih16,   (size_t)H_DIM * IN_DIM);
    cvt(W_hh,   Whh16,   (size_t)H_DIM * H_DIM);
    cvt(W_head, Whead16, (size_t)OUT_DIM * H_DIM);

    dim3 blk(256);
    dim3 grdH(H_DIM / BN, B_DIM / BM);       // (4, 64)
    rnn_xp_kernel<<<grdH, blk, 0, stream>>>(x16, Wih16, b_ih, b_hh, xp, hA);

    const _Float16* hs = hA;
    _Float16* hd = hB;
    for (int s = 1; s < steps; ++s) {        // 31 recurrent steps (h1 already done)
        rnn_step_kernel<<<grdH, blk, 0, stream>>>(hs, Whh16, xp, hd);
        const _Float16* t = hd; hd = (_Float16*)hs; hs = t;
    }

    dim3 grdO(OUT_DIM / BN, B_DIM / BM);     // (2, 64)
    rnn_head_kernel<<<grdO, blk, 0, stream>>>(hs, Whead16, b_head, (float*)d_out);
}
